// MOMENTPretrain_90074054131738
// MI455X (gfx1250) — matrix operand order repaired and checked by reading, not compiled, not run on hardware
//
#include <hip/hip_runtime.h>
#include <hip/hip_bf16.h>
#include <math.h>

// ---------------- constants ----------------
#define D_MODEL 768
#define N_LAYERS 12
#define N_HEADS 12
#define D_KV 64
#define D_FF 3072
#define PATCH 8
#define N_PATCH 64
#define BATCH 128
#define ROWS (BATCH * N_PATCH)            // 8192 token rows
#define EPSV 1e-6f

// ---------------- WMMA types ----------------
typedef __attribute__((ext_vector_type(16))) __bf16 v16bf;
typedef __attribute__((ext_vector_type(8)))  float  v8f;

// Fragment built from two 16-byte dword loads; bf16 pairs in memory map 1:1
// onto the per-VGPR packing of the WMMA operand layout -> no unpack VALU.
union Frag {
    uint4        q[2];
    unsigned int d[8];
    v16bf        v;
};

__device__ __forceinline__ unsigned short f2bf(float f) {
    unsigned int x = __float_as_uint(f);
    unsigned int r = (x + 0x7FFFu + ((x >> 16) & 1u)) >> 16;   // RNE
    return (unsigned short)r;
}
__device__ __forceinline__ float bf2f(unsigned short u) {
    return __uint_as_float((unsigned int)u << 16);
}

// =====================================================================
// bf16 WMMA GEMM: Out[M,N] = epilogue( A[M,K](bf16) @ Wt[N,K](bf16)^T )
// One wave computes a 32x64 tile: 2x4 grid of 16x16x32 bf16 WMMAs.
// Per K-step: 12 x global_load_b128 + 8 x v_wmma.
//   A layout  (16-bit A 16x32): lane l16=row, dwords {0..3}+4g, {8..11}+4g
//   B layout  (16-bit B 32x16): lane l16=col, dwords {0..3}+4g, {8..11}+4g (Wt is [N,K])
// =====================================================================
template <bool RESIDUAL, bool RELU, bool OUT_BF16>
__global__ void gemm_bf16_wmma(const unsigned short* __restrict__ A,
                               const unsigned short* __restrict__ Wt,
                               const float* __restrict__ Res,
                               void* __restrict__ Out,
                               int M, int K, int N) {
    const int lane   = threadIdx.x & 31;
    const int wib    = threadIdx.x >> 5;
    const int wavesB = blockDim.x >> 5;
    const int waveId = blockIdx.x * wavesB + wib;
    const int tilesN = N >> 6;                    // 64-wide tiles
    const int tilesM = M >> 5;                    // 32-tall tiles
    if (waveId >= tilesM * tilesN) return;        // wave-uniform branch

    const int tM = waveId / tilesN;
    const int tN = waveId % tilesN;
    const int mBase = tM << 5;
    const int nBase = tN << 6;
    const int l16 = lane & 15;
    const int g   = lane >> 4;
    const int K2  = K >> 1;                       // dwords per row

    const unsigned int* A32 = (const unsigned int*)A;
    const unsigned int* W32 = (const unsigned int*)Wt;

    // per-lane row/col base pointers (dword units)
    const unsigned int* pA[2];
#pragma unroll
    for (int t = 0; t < 2; ++t)
        pA[t] = A32 + (size_t)(mBase + (t << 4) + l16) * K2 + 4 * g;
    const unsigned int* pB[4];
#pragma unroll
    for (int t = 0; t < 4; ++t)
        pB[t] = W32 + (size_t)(nBase + (t << 4) + l16) * K2 + 4 * g;

    v8f acc[2][4] = {};

    for (int kb2 = 0; kb2 < K2; kb2 += 16) {      // 16 dwords = K-step of 32
        Frag a[2], b[4];
#pragma unroll
        for (int t = 0; t < 2; ++t) {
            a[t].q[0] = *(const uint4*)(pA[t] + kb2);
            a[t].q[1] = *(const uint4*)(pA[t] + kb2 + 8);
        }
#pragma unroll
        for (int t = 0; t < 4; ++t) {
            b[t].q[0] = *(const uint4*)(pB[t] + kb2);
            b[t].q[1] = *(const uint4*)(pB[t] + kb2 + 8);
        }
#pragma unroll
        for (int i = 0; i < 2; ++i)
#pragma unroll
            for (int j = 0; j < 4; ++j)
                acc[i][j] = __builtin_amdgcn_wmma_f32_16x16x32_bf16(
                    false, a[i].v, false, b[j].v, (short)0, acc[i][j], false, false);
    }

    // epilogue: C/D layout n = lane%16, m = r + 8*(lane/16)
#pragma unroll
    for (int i = 0; i < 2; ++i) {
#pragma unroll
        for (int j = 0; j < 4; ++j) {
            const int n = nBase + (j << 4) + l16;
#pragma unroll
            for (int r = 0; r < 8; ++r) {
                const int m = mBase + (i << 4) + r + 8 * g;
                const size_t off = (size_t)m * N + n;
                float val = acc[i][j][r];
                if constexpr (RESIDUAL) val += Res[off];
                if constexpr (RELU) val = val > 0.0f ? val : 0.0f;
                if constexpr (OUT_BF16)
                    ((unsigned short*)Out)[off] = f2bf(val);
                else
                    ((float*)Out)[off] = val;
            }
        }
    }
}

// =====================================================================
// Weight convert + transpose: W[K,N] fp32 -> Wt[N,K] bf16
// =====================================================================
__global__ void convT_kernel(const float* __restrict__ W,
                             unsigned short* __restrict__ Wt,
                             int K, int N) {
    const int idx = blockIdx.x * blockDim.x + threadIdx.x;
    if (idx >= K * N) return;
    const int n = idx / K;
    const int k = idx - n * K;
    Wt[idx] = f2bf(W[(size_t)k * N + n]);
}

// =====================================================================
// Patch embedding + mask-token substitution: h[b,n,d] (fp32)
// =====================================================================
__global__ void patch_embed_kernel(const float* __restrict__ x,
                                   const unsigned char* __restrict__ mask,
                                   const float* __restrict__ Wp,
                                   const float* __restrict__ bp,
                                   const float* __restrict__ mtok,
                                   float* __restrict__ h) {
    const int idx = blockIdx.x * blockDim.x + threadIdx.x;
    if (idx >= ROWS * D_MODEL) return;
    const int d  = idx % D_MODEL;
    const int bn = idx / D_MODEL;
    if (mask[bn]) { h[idx] = mtok[d]; return; }
    const float* xr = x + (size_t)bn * PATCH;
    float acc = bp[d];
#pragma unroll
    for (int p = 0; p < PATCH; ++p) acc = fmaf(xr[p], Wp[p * D_MODEL + d], acc);
    h[idx] = acc;
}

// =====================================================================
// T5 relative position bias -> bias[H, N, N]
// =====================================================================
__global__ void rel_bias_kernel(const float* __restrict__ emb,
                                float* __restrict__ bias) {
    const int idx = blockIdx.x * blockDim.x + threadIdx.x;
    if (idx >= N_HEADS * N_PATCH * N_PATCH) return;
    const int k = idx & 63;
    const int q = (idx >> 6) & 63;
    const int h = idx >> 12;
    const int rel = k - q;                         // pos[None,:] - pos[:,None]
    int ret = (rel > 0) ? 16 : 0;                  // num_buckets/2
    int n = rel < 0 ? -rel : rel;
    int bucket;
    if (n < 8) {                                   // max_exact = 8
        bucket = ret + n;
    } else {
        int vl = 8 + (int)(logf((float)n / 8.0f) / logf(16.0f) * 8.0f);
        if (vl > 15) vl = 15;
        bucket = ret + vl;
    }
    bias[idx] = emb[bucket * N_HEADS + h];
}

// =====================================================================
// RMSNorm (T5): one block per row; OUT_BF16 selects output precision
// =====================================================================
template <bool OUT_BF16>
__global__ void rmsnorm_kernel(const float* __restrict__ x,
                               const float* __restrict__ w,
                               void* __restrict__ out) {
    const int row = blockIdx.x;
    const float* xr = x + (size_t)row * D_MODEL;
    __shared__ float red[256];
    float ss = 0.0f;
    for (int i = threadIdx.x; i < D_MODEL; i += 256) {
        const float v = xr[i];
        ss = fmaf(v, v, ss);
    }
    red[threadIdx.x] = ss;
    __syncthreads();
    for (int s = 128; s > 0; s >>= 1) {
        if (threadIdx.x < s) red[threadIdx.x] += red[threadIdx.x + s];
        __syncthreads();
    }
    const float inv = rsqrtf(red[0] / (float)D_MODEL + EPSV);
    for (int i = threadIdx.x; i < D_MODEL; i += 256) {
        const float v = xr[i] * inv * w[i];
        if constexpr (OUT_BF16)
            ((unsigned short*)out)[(size_t)row * D_MODEL + i] = f2bf(v);
        else
            ((float*)out)[(size_t)row * D_MODEL + i] = v;
    }
}

// =====================================================================
// Attention per (batch, head): scores=q.k^T + bias, softmax, ctx=attn@v
// q/k/v/ctx are bf16 in global; all math fp32 in LDS.
// =====================================================================
__global__ void attention_kernel(const unsigned short* __restrict__ q,
                                 const unsigned short* __restrict__ k,
                                 const unsigned short* __restrict__ v,
                                 const float* __restrict__ bias,
                                 unsigned short* __restrict__ ctx) {
    const int bh = blockIdx.x;
    const int b  = bh / N_HEADS;
    const int hh = bh % N_HEADS;
    __shared__ float smA[N_PATCH * D_KV];   // q, later v
    __shared__ float smK[N_PATCH * D_KV];
    __shared__ float smS[N_PATCH * N_PATCH];
    const size_t base = (size_t)b * N_PATCH * D_MODEL + hh * D_KV;

    for (int i = threadIdx.x; i < N_PATCH * D_KV; i += 256) {
        const int r = i >> 6, c = i & 63;
        smA[i] = bf2f(q[base + (size_t)r * D_MODEL + c]);
        smK[i] = bf2f(k[base + (size_t)r * D_MODEL + c]);
    }
    __syncthreads();

    const float* bslice = bias + (size_t)hh * N_PATCH * N_PATCH;
    for (int i = threadIdx.x; i < N_PATCH * N_PATCH; i += 256) {
        const int qr = i >> 6, kr = i & 63;
        float acc = bslice[i];                     // T5: no 1/sqrt(d) scale
#pragma unroll 8
        for (int d = 0; d < D_KV; ++d)
            acc = fmaf(smA[qr * D_KV + d], smK[kr * D_KV + d], acc);
        smS[i] = acc;
    }
    __syncthreads();

    // load v over q's LDS while threads 0..63 run row softmax
    for (int i = threadIdx.x; i < N_PATCH * D_KV; i += 256) {
        const int r = i >> 6, c = i & 63;
        smA[i] = bf2f(v[base + (size_t)r * D_MODEL + c]);
    }
    if (threadIdx.x < N_PATCH) {
        float* row = smS + threadIdx.x * N_PATCH;
        float mx = row[0];
        for (int j = 1; j < N_PATCH; ++j) mx = fmaxf(mx, row[j]);
        float sum = 0.0f;
        for (int j = 0; j < N_PATCH; ++j) {
            const float e = __expf(row[j] - mx);
            row[j] = e;
            sum += e;
        }
        const float inv = 1.0f / sum;
        for (int j = 0; j < N_PATCH; ++j) row[j] *= inv;
    }
    __syncthreads();

    for (int i = threadIdx.x; i < N_PATCH * D_KV; i += 256) {
        const int qr = i >> 6, d = i & 63;
        float acc = 0.0f;
#pragma unroll 8
        for (int kk = 0; kk < N_PATCH; ++kk)
            acc = fmaf(smS[qr * N_PATCH + kk], smA[kk * D_KV + d], acc);
        ctx[base + (size_t)qr * D_MODEL + d] = f2bf(acc);
    }
}

// =====================================================================
// Head projection: pred[b,n,p] = hn . head_W[:,p] + head_b[p]   (fp32)
// =====================================================================
__global__ void head_kernel(const float* __restrict__ hn,
                            const float* __restrict__ Wh,
                            const float* __restrict__ bh,
                            float* __restrict__ pred) {
    const int idx = blockIdx.x * blockDim.x + threadIdx.x;
    if (idx >= ROWS * PATCH) return;
    const int p   = idx & 7;
    const int row = idx >> 3;
    const float* hr = hn + (size_t)row * D_MODEL;
    float acc = bh[p];
    for (int d = 0; d < D_MODEL; ++d) acc = fmaf(hr[d], Wh[d * PATCH + p], acc);
    pred[idx] = acc;
}

// =====================================================================
// Masked MSE, deterministic single-block tree reduction
// =====================================================================
__global__ void loss_kernel(const float* __restrict__ pred,
                            const float* __restrict__ x,
                            const unsigned char* __restrict__ mask,
                            float* __restrict__ out) {
    __shared__ float rs[256], rc[256];
    float s = 0.0f, c = 0.0f;
    for (int bn = threadIdx.x; bn < ROWS; bn += 256) {
        if (mask[bn]) {
            c += (float)PATCH;
            const float* pr = pred + (size_t)bn * PATCH;
            const float* xr = x + (size_t)bn * PATCH;
#pragma unroll
            for (int p = 0; p < PATCH; ++p) {
                const float d = pr[p] - xr[p];
                s = fmaf(d, d, s);
            }
        }
    }
    rs[threadIdx.x] = s;
    rc[threadIdx.x] = c;
    __syncthreads();
    for (int st = 128; st > 0; st >>= 1) {
        if (threadIdx.x < st) {
            rs[threadIdx.x] += rs[threadIdx.x + st];
            rc[threadIdx.x] += rc[threadIdx.x + st];
        }
        __syncthreads();
    }
    if (threadIdx.x == 0) out[0] = rs[0] / fmaxf(rc[0], 1.0f);
}

// =====================================================================
// Host-side orchestration
// =====================================================================
static inline int gemm_grid(int M, int N) {
    const int waves = (M >> 5) * (N >> 6);        // 32x64 tile per wave
    return (waves + 7) / 8;                       // 8 waves (256 threads)/block
}

extern "C" void kernel_launch(void* const* d_in, const int* in_sizes, int n_in,
                              void* d_out, int out_size, void* d_ws, size_t ws_size,
                              hipStream_t stream) {
    (void)in_sizes; (void)n_in; (void)out_size; (void)ws_size;

    const float*         x        = (const float*)d_in[0];
    const unsigned char* mask     = (const unsigned char*)d_in[1];
    const float*         patch_W  = (const float*)d_in[2];
    const float*         patch_b  = (const float*)d_in[3];
    const float*         mtok     = (const float*)d_in[4];
    const float*         rel_emb  = (const float*)d_in[5];
    const float*         ln1      = (const float*)d_in[6];
    const float*         q_W      = (const float*)d_in[7];
    const float*         k_W      = (const float*)d_in[8];
    const float*         v_W      = (const float*)d_in[9];
    const float*         o_W      = (const float*)d_in[10];
    const float*         ln2      = (const float*)d_in[11];
    const float*         wi_W     = (const float*)d_in[12];
    const float*         wo_W     = (const float*)d_in[13];
    const float*         final_ln = (const float*)d_in[14];
    const float*         head_W   = (const float*)d_in[15];
    const float*         head_b   = (const float*)d_in[16];

    float* out  = (float*)d_out;                  // [0]=loss, [1..]=pred
    float* pred = out + 1;

    const size_t SZ = (size_t)ROWS * D_MODEL;     // 6.29M elements

    // workspace carve-up (all 256B aligned)
    char* p = (char*)d_ws;
    auto take = [&p](size_t bytes) {
        char* r = p;
        p += (bytes + 255) & ~(size_t)255;
        return r;
    };
    float*          h    = (float*)take(SZ * 4);
    float*          hnf  = (float*)take(SZ * 4);             // fp32 final norm
    float*          bias = (float*)take((size_t)N_HEADS * N_PATCH * N_PATCH * 4);
    unsigned short* hnb  = (unsigned short*)take(SZ * 2);    // bf16 normed acts
    unsigned short* qb   = (unsigned short*)take(SZ * 2);
    unsigned short* kb   = (unsigned short*)take(SZ * 2);
    unsigned short* vb   = (unsigned short*)take(SZ * 2);
    unsigned short* ctxb = (unsigned short*)take(SZ * 2);
    unsigned short* ff1  = (unsigned short*)take((size_t)ROWS * D_FF * 2);
    unsigned short* wqT  = (unsigned short*)take((size_t)D_MODEL * D_MODEL * 2);
    unsigned short* wkT  = (unsigned short*)take((size_t)D_MODEL * D_MODEL * 2);
    unsigned short* wvT  = (unsigned short*)take((size_t)D_MODEL * D_MODEL * 2);
    unsigned short* woT  = (unsigned short*)take((size_t)D_MODEL * D_MODEL * 2);
    unsigned short* wiT  = (unsigned short*)take((size_t)D_MODEL * D_FF * 2);
    unsigned short* wo2T = (unsigned short*)take((size_t)D_FF * D_MODEL * 2);

    const int T = 256;
    const int NSQ = D_MODEL * D_MODEL;            // 589824
    const int NFF = D_MODEL * D_FF;               // 2359296

    patch_embed_kernel<<<(int)((SZ + T - 1) / T), T, 0, stream>>>(
        x, mask, patch_W, patch_b, mtok, h);
    rel_bias_kernel<<<(N_HEADS * N_PATCH * N_PATCH + T - 1) / T, T, 0, stream>>>(
        rel_emb, bias);

    for (int l = 0; l < N_LAYERS; ++l) {
        const float* l1  = ln1  + (size_t)l * D_MODEL;
        const float* l2  = ln2  + (size_t)l * D_MODEL;
        const float* qw  = q_W  + (size_t)l * NSQ;
        const float* kw  = k_W  + (size_t)l * NSQ;
        const float* vw  = v_W  + (size_t)l * NSQ;
        const float* ow  = o_W  + (size_t)l * NSQ;
        const float* wiw = wi_W + (size_t)l * NFF;
        const float* wow = wo_W + (size_t)l * NFF;

        // weight convert+transpose (bf16, [N,K] layout)
        convT_kernel<<<(NSQ + T - 1) / T, T, 0, stream>>>(qw, wqT, D_MODEL, D_MODEL);
        convT_kernel<<<(NSQ + T - 1) / T, T, 0, stream>>>(kw, wkT, D_MODEL, D_MODEL);
        convT_kernel<<<(NSQ + T - 1) / T, T, 0, stream>>>(vw, wvT, D_MODEL, D_MODEL);
        convT_kernel<<<(NSQ + T - 1) / T, T, 0, stream>>>(ow, woT, D_MODEL, D_MODEL);
        convT_kernel<<<(NFF + T - 1) / T, T, 0, stream>>>(wiw, wiT, D_MODEL, D_FF);
        convT_kernel<<<(NFF + T - 1) / T, T, 0, stream>>>(wow, wo2T, D_FF, D_MODEL);

        rmsnorm_kernel<true><<<ROWS, T, 0, stream>>>(h, l1, hnb);

        gemm_bf16_wmma<false, false, true><<<gemm_grid(ROWS, D_MODEL), T, 0, stream>>>(
            hnb, wqT, nullptr, qb, ROWS, D_MODEL, D_MODEL);
        gemm_bf16_wmma<false, false, true><<<gemm_grid(ROWS, D_MODEL), T, 0, stream>>>(
            hnb, wkT, nullptr, kb, ROWS, D_MODEL, D_MODEL);
        gemm_bf16_wmma<false, false, true><<<gemm_grid(ROWS, D_MODEL), T, 0, stream>>>(
            hnb, wvT, nullptr, vb, ROWS, D_MODEL, D_MODEL);

        attention_kernel<<<BATCH * N_HEADS, T, 0, stream>>>(qb, kb, vb, bias, ctxb);

        // h = h + ctx @ oW   (residual fused, fp32 out)
        gemm_bf16_wmma<true, false, false><<<gemm_grid(ROWS, D_MODEL), T, 0, stream>>>(
            ctxb, woT, h, h, ROWS, D_MODEL, D_MODEL);

        rmsnorm_kernel<true><<<ROWS, T, 0, stream>>>(h, l2, hnb);

        // ff1 = relu(hn @ wiW) stored bf16
        gemm_bf16_wmma<false, true, true><<<gemm_grid(ROWS, D_FF), T, 0, stream>>>(
            hnb, wiT, nullptr, ff1, ROWS, D_MODEL, D_FF);

        // h = h + ff1 @ woW  (residual fused, fp32 out)
        gemm_bf16_wmma<true, false, false><<<gemm_grid(ROWS, D_MODEL), T, 0, stream>>>(
            ff1, wo2T, h, h, ROWS, D_FF, D_MODEL);
    }

    rmsnorm_kernel<false><<<ROWS, T, 0, stream>>>(h, final_ln, hnf);
    head_kernel<<<(ROWS * PATCH + T - 1) / T, T, 0, stream>>>(hnf, head_W, head_b, pred);
    loss_kernel<<<1, T, 0, stream>>>(pred, x, mask, out);
}
